// Net2_7395933683786
// MI455X (gfx1250) — matrix operand order repaired and checked by reading, not compiled, not run on hardware
//
#include <hip/hip_runtime.h>
#include <math.h>

typedef __bf16 bf16_t;
typedef __bf16 v16bf __attribute__((ext_vector_type(16)));
typedef float  v8f   __attribute__((ext_vector_type(8)));

#define DEV __device__ __forceinline__

struct U8x4 { uint4 a, b; };   // 32B carrier for one v16bf fragment

DEV unsigned pk_bf16(float a, float b) {
  unsigned short x = __builtin_bit_cast(unsigned short, (bf16_t)a);
  unsigned short y = __builtin_bit_cast(unsigned short, (bf16_t)b);
  return (unsigned)x | ((unsigned)y << 16);
}

DEV void fma4(float4& a, const float4 v, float m) {
  a.x += v.x * m; a.y += v.y * m; a.z += v.z * m; a.w += v.w * m;
}
DEV float4 scl4(const float4 v, float m) {
  float4 r; r.x = v.x * m; r.y = v.y * m; r.z = v.z * m; r.w = v.w * m; return r;
}

// ---------------------------------------------------------------------------
// WMMA GEMM: C[M,N] = act(A1[M,K] @ W1 (+ A2 @ W2) + bias).
// A: f32 row-major, converted to bf16 while staging to LDS in
// fragment-native order. W: pre-packed (pack_w) fragment-native bf16 tiles,
// staged with a contiguous b128 copy. Double-buffered LDS, 1 barrier/step.
// Block: 256 thr (8 waves), tile 128x128, K-step 32, 8 WMMAs/wave/step.
// ---------------------------------------------------------------------------
__global__ __launch_bounds__(256) void gemm_wmma(
    const float* __restrict__ A1, const unsigned* __restrict__ W1p,
    const float* __restrict__ A2, const unsigned* __restrict__ W2p,
    const float* __restrict__ bias, float* __restrict__ C,
    int M, int N, int K, int relu)
{
  __shared__ unsigned AslF[2][2048];   // [mt 0..7][half][lane][4] DWORDs
  __shared__ unsigned BslF[2][2048];   // [nt 0..7][half][lane][4] DWORDs

  const int tid  = threadIdx.x;
  const int lane = tid & 31;
  const int wave = tid >> 5;
  const int wm = wave >> 1;           // 0..3 -> 32-row strip
  const int wn = wave & 1;            // 0..1 -> 64-col strip
  const int kh = lane >> 4;
  const int ml = lane & 15;

  const int m0 = blockIdx.y * 128;
  const int n0 = blockIdx.x * 128;
  const int ntilesN = (N + 127) >> 7;
  const int kiters  = K >> 5;
  const int nsrc    = (A2 != nullptr) ? 2 : 1;
  const int steps   = nsrc * kiters;

  v8f acc[2][4] = {};

  auto stage = [&](int st, int b) {
    const int s  = (st >= kiters) ? 1 : 0;
    const int kt = (st - (s ? kiters : 0)) << 5;
    const float*    A  = s ? A2 : A1;
    const unsigned* Wp = s ? W2p : W1p;
    // ---- A tile: 128 rows x 32 k, f32 -> packed bf16 pairs, frag-native
    {
      const int r    = tid >> 1;
      const int half = tid & 1;
      const int mt   = r >> 4;
      const int ml15 = r & 15;
      const float* Ar = A + (size_t)(m0 + r) * K + kt + half * 16;
      unsigned* Ad = &AslF[b][0];
      #pragma unroll
      for (int q = 0; q < 4; ++q) {
        float4 f = ((const float4*)Ar)[q];
        const int kp0 = half * 8 + 2 * q;
        {
          const int kp = kp0;
          const int khh = kp >> 3;
          const int v   = kp & 7;
          Ad[mt * 256 + (v >> 2) * 128 + (khh * 16 + ml15) * 4 + (v & 3)] = pk_bf16(f.x, f.y);
        }
        {
          const int kp = kp0 + 1;
          const int khh = kp >> 3;
          const int v   = kp & 7;
          Ad[mt * 256 + (v >> 2) * 128 + (khh * 16 + ml15) * 4 + (v & 3)] = pk_bf16(f.z, f.w);
        }
      }
    }
    // ---- B tile: straight contiguous copy of a pre-packed 8KB tile
    {
      const uint4* t4 = (const uint4*)(Wp + ((size_t)(kt >> 5) * ntilesN + (n0 >> 7)) * 2048);
      uint4* d4 = (uint4*)&BslF[b][0];
      d4[tid]       = t4[tid];
      d4[tid + 256] = t4[tid + 256];
    }
  };

  stage(0, 0);
  __syncthreads();
  for (int st = 0; st < steps; ++st) {
    const int cur = st & 1;
    if (st + 1 < steps) stage(st + 1, cur ^ 1);

    const unsigned* Ab = &AslF[cur][0];
    const unsigned* Bb = &BslF[cur][0];
    v16bf af[2], bfr[4];
    #pragma unroll
    for (int i = 0; i < 2; ++i) {
      const uint4* pa = (const uint4*)(Ab + (wm * 2 + i) * 256);
      U8x4 u; u.a = pa[lane]; u.b = pa[lane + 32];
      af[i] = __builtin_bit_cast(v16bf, u);
    }
    #pragma unroll
    for (int j = 0; j < 4; ++j) {
      const uint4* pb = (const uint4*)(Bb + (wn * 4 + j) * 256);
      U8x4 u; u.a = pb[lane]; u.b = pb[lane + 32];
      bfr[j] = __builtin_bit_cast(v16bf, u);
    }
    #pragma unroll
    for (int i = 0; i < 2; ++i)
      #pragma unroll
      for (int j = 0; j < 4; ++j)
        acc[i][j] = __builtin_amdgcn_wmma_f32_16x16x32_bf16(
            false, af[i], false, bfr[j], (short)0, acc[i][j], false, false);
    __syncthreads();
  }

  // ---- epilogue: bias + relu + store (C/D layout: M = r + 8*kh, N = ml)
  #pragma unroll
  for (int i = 0; i < 2; ++i) {
    #pragma unroll
    for (int r = 0; r < 8; ++r) {
      const int mrow = m0 + wm * 32 + i * 16 + kh * 8 + r;
      #pragma unroll
      for (int j = 0; j < 4; ++j) {
        const int ncol = n0 + wn * 64 + j * 16 + ml;
        if (ncol < N) {
          float v = acc[i][j][r];
          if (bias) v += bias[ncol];
          if (relu) v = fmaxf(v, 0.0f);
          C[(size_t)mrow * N + ncol] = v;
        }
      }
    }
  }
}

// ---------------------------------------------------------------------------
// Pack f32 weights [K,N] into WMMA-fragment-native bf16 tiles:
// tile (ktile, ntile) = 2048 DWORDs, pos = nt*256 + (v>>2)*128 + lane*4 +
// (v&3), DWORD = {W[k][n], W[k+1][n]} with kp = kh*8+v, k = ktile*32+2*kp,
// n = ntile*128 + nt*16 + (lane&15), kh = lane>>4. Zero-pads n >= N.
// ---------------------------------------------------------------------------
__global__ __launch_bounds__(256) void pack_w(const float* __restrict__ W,
                                              unsigned* __restrict__ Wp,
                                              int K, int N, int ntiles)
{
  const int idx = blockIdx.x * 256 + threadIdx.x;
  const int total = (K >> 5) * ntiles * 2048;
  if (idx >= total) return;
  const int pos   = idx & 2047;
  const int tile  = idx >> 11;
  const int ntile = tile % ntiles;
  const int ktile = tile / ntiles;
  const int nt    = pos >> 8;
  const int rem   = pos & 255;
  const int half  = rem >> 7;
  const int li    = (rem >> 2) & 31;
  const int vlow  = rem & 3;
  const int khh   = li >> 4;
  const int v     = half * 4 + vlow;
  const int kp    = khh * 8 + v;
  const int k     = (ktile << 5) + (kp << 1);
  const int n     = (ntile << 7) + (nt << 4) + (li & 15);
  float a = 0.0f, b = 0.0f;
  if (n < N) {
    a = W[(size_t)k * N + n];
    b = W[(size_t)(k + 1) * N + n];
  }
  Wp[idx] = pk_bf16(a, b);
}

// ---------------------------------------------------------------------------
// SAGE mean-aggregate: block per graph (512 edges in LDS), wave per node,
// ballot-compacted edge scan, deterministic ascending edge order.
// ---------------------------------------------------------------------------
__global__ __launch_bounds__(256) void sage_agg(
    const float* __restrict__ x,
    const int* __restrict__ esrc, const int* __restrict__ edst,
    const float* __restrict__ emask,
    int n_per, float* __restrict__ agg)
{
  __shared__ int   sSrc[512];
  __shared__ int   sDst[512];
  __shared__ float sMk[512];
  const int g    = blockIdx.x;
  const int tid  = threadIdx.x;
  const int lane = tid & 31;
  const int wave = tid >> 5;
  for (int e = tid; e < 512; e += 256) {
    int ge = g * 512 + e;
    sSrc[e] = esrc[ge];
    sDst[e] = edst[ge];
    sMk[e]  = emask ? emask[ge] : 1.0f;
  }
  __syncthreads();
  const float4* x4 = (const float4*)x;
  for (int n = wave; n < n_per; n += 8) {
    const int gnode = g * n_per + n;
    float4 a0 = {0,0,0,0}, a1 = {0,0,0,0}, a2 = {0,0,0,0}, a3 = {0,0,0,0};
    float cnt = 0.0f;
    for (int base = 0; base < 512; base += 32) {
      const int e = base + lane;
      bool match = (sDst[e] == gnode) && (sMk[e] > 0.0f);
      unsigned long long mm = __ballot((int)match);
      while (mm) {
        int b = __builtin_ctzll(mm);
        mm &= mm - 1;
        int eb = base + b;
        int srow = sSrc[eb];
        float mk = sMk[eb];
        const float4* row = x4 + (size_t)srow * 128;
        fma4(a0, row[lane +  0], mk);
        fma4(a1, row[lane + 32], mk);
        fma4(a2, row[lane + 64], mk);
        fma4(a3, row[lane + 96], mk);
        cnt += mk;
      }
    }
    float rc = 1.0f / fmaxf(cnt, 1.0f);
    float4* o = (float4*)agg + (size_t)gnode * 128;
    o[lane +  0] = scl4(a0, rc);
    o[lane + 32] = scl4(a1, rc);
    o[lane + 64] = scl4(a2, rc);
    o[lane + 96] = scl4(a3, rc);
  }
}

// ---------------------------------------------------------------------------
// TopK pooling: block per graph. score=tanh(x.p/||p||), stable descending
// rank (ties -> lower index), gate+gather, edge remap with emask.
// ---------------------------------------------------------------------------
__global__ __launch_bounds__(256) void topk_pool(
    const float* __restrict__ x, int n_per, int k,
    const float* __restrict__ p, const float* __restrict__ pnorm,
    const int* __restrict__ src_in, const int* __restrict__ dst_in,
    const float* __restrict__ em_in,
    float* __restrict__ xout, int* __restrict__ src_out, int* __restrict__ dst_out,
    float* __restrict__ em_out)
{
  __shared__ float sP[512];
  __shared__ float sScore[64];
  __shared__ int   sRank[64];
  const int g    = blockIdx.x;
  const int tid  = threadIdx.x;
  const int lane = tid & 31;
  const int wave = tid >> 5;
  for (int d = tid; d < 512; d += 256) sP[d] = p[d];
  __syncthreads();
  const float nrm = pnorm[0];
  const float4* x4 = (const float4*)x;
  const float4* p4 = (const float4*)sP;
  for (int n = wave; n < n_per; n += 8) {
    const float4* row = x4 + (size_t)(g * n_per + n) * 128;
    float d = 0.0f;
    #pragma unroll
    for (int t = 0; t < 4; ++t) {
      float4 xv = row[lane + 32 * t];
      float4 pv = p4[lane + 32 * t];
      d += xv.x * pv.x + xv.y * pv.y + xv.z * pv.z + xv.w * pv.w;
    }
    #pragma unroll
    for (int off = 16; off > 0; off >>= 1) d += __shfl_xor(d, off);
    if (lane == 0) sScore[n] = tanhf(d / nrm);
  }
  __syncthreads();
  if (tid < n_per) {
    float s = sScore[tid];
    int r = 0;
    for (int j = 0; j < n_per; ++j) {
      float sj = sScore[j];
      r += (int)((sj > s) || (sj == s && j < tid));
    }
    sRank[tid] = (r < k) ? r : -1;
  }
  __syncthreads();
  for (int n = wave; n < n_per; n += 8) {
    int r = sRank[n];
    if (r >= 0) {
      float sc = sScore[n];
      const float4* row = x4 + (size_t)(g * n_per + n) * 128;
      float4* orow = (float4*)xout + (size_t)(g * k + r) * 128;
      #pragma unroll
      for (int t = 0; t < 4; ++t) orow[lane + 32 * t] = scl4(row[lane + 32 * t], sc);
    }
  }
  for (int e = tid; e < 512; e += 256) {
    int ge = g * 512 + e;
    float m = em_in ? em_in[ge] : 1.0f;
    int ns = 0, nd = 0; float nm = 0.0f;
    if (m > 0.0f) {
      int sl = src_in[ge] - g * n_per;
      int dl = dst_in[ge] - g * n_per;
      int rs = sRank[sl];
      int rd = sRank[dl];
      if (rs >= 0 && rd >= 0) { ns = g * k + rs; nd = g * k + rd; nm = 1.0f; }
    }
    src_out[ge] = ns; dst_out[ge] = nd; em_out[ge] = nm;
  }
}

// readout: h[g] (+)= concat(max over k rows, mean over k rows)
__global__ __launch_bounds__(256) void readout(const float* __restrict__ xp, int k,
                                               float* __restrict__ h, int accum)
{
  const int g = blockIdx.x;
  for (int d = threadIdx.x; d < 512; d += 256) {
    float mx = -3.402823466e38f, sm = 0.0f;
    for (int r = 0; r < k; ++r) {
      float v = xp[(size_t)(g * k + r) * 512 + d];
      mx = fmaxf(mx, v); sm += v;
    }
    float mean = sm / (float)k;
    size_t o = (size_t)g * 1024;
    if (accum) { h[o + d] += mx; h[o + 512 + d] += mean; }
    else       { h[o + d]  = mx; h[o + 512 + d]  = mean; }
  }
}

__global__ __launch_bounds__(128) void embed_gather(const int* __restrict__ ids,
    const float* __restrict__ emb, float* __restrict__ x)
{
  const int row = blockIdx.x;
  const int id  = ids[row];
  const float4* s = (const float4*)emb + (size_t)id * 128;
  float4* d = (float4*)x + (size_t)row * 128;
  d[threadIdx.x] = s[threadIdx.x];
}

__global__ __launch_bounds__(256) void vec_norm(const float* __restrict__ p,
                                                float* __restrict__ out)
{
  __shared__ float red[256];
  int t = threadIdx.x;
  float a = p[t], b = p[t + 256];
  red[t] = a * a + b * b;
  __syncthreads();
  for (int s = 128; s > 0; s >>= 1) { if (t < s) red[t] += red[t + s]; __syncthreads(); }
  if (t == 0) out[0] = sqrtf(red[0]);
}

__global__ __launch_bounds__(256) void mlp_final(const float* __restrict__ h4,
    const float* __restrict__ w, const float* __restrict__ b, float* __restrict__ out)
{
  const int lane = threadIdx.x & 31;
  const int bi   = blockIdx.x * 8 + (threadIdx.x >> 5);
  float s = h4[(size_t)bi * 64 + lane] * w[lane]
          + h4[(size_t)bi * 64 + 32 + lane] * w[32 + lane];
  #pragma unroll
  for (int off = 16; off > 0; off >>= 1) s += __shfl_xor(s, off);
  if (lane == 0) out[bi] = 1.0f / (1.0f + expf(-(s + b[0])));
}

extern "C" void kernel_launch(void* const* d_in, const int* in_sizes, int n_in,
                              void* d_out, int out_size, void* d_ws, size_t ws_size,
                              hipStream_t stream)
{
  const int*   x_ids = (const int*)d_in[0];
  const int*   e_src = (const int*)d_in[1];
  const int*   e_dst = (const int*)d_in[2];
  const float* emb   = (const float*)d_in[3];
  const float* w1l = (const float*)d_in[4];  const float* b1l = (const float*)d_in[5];  const float* w1r = (const float*)d_in[6];
  const float* w2l = (const float*)d_in[7];  const float* b2l = (const float*)d_in[8];  const float* w2r = (const float*)d_in[9];
  const float* w3l = (const float*)d_in[10]; const float* b3l = (const float*)d_in[11]; const float* w3r = (const float*)d_in[12];
  const float* p1  = (const float*)d_in[13]; const float* p2  = (const float*)d_in[14]; const float* p3 = (const float*)d_in[15];
  const float* lw1 = (const float*)d_in[16]; const float* lb1 = (const float*)d_in[17];
  const float* lw2 = (const float*)d_in[18]; const float* lb2 = (const float*)d_in[19];
  const float* lw3 = (const float*)d_in[20]; const float* lb3 = (const float*)d_in[21];
  const float* lw4 = (const float*)d_in[22]; const float* lb4 = (const float*)d_in[23];
  const float* lw5 = (const float*)d_in[24]; const float* lb5 = (const float*)d_in[25];
  (void)in_sizes; (void)n_in; (void)out_size; (void)ws_size;

  char* base = (char*)d_ws;
  size_t off = 0;
  auto carve = [&](size_t bytes) -> void* {
    void* p = base + off;
    off += (bytes + 255) & ~(size_t)255;
    return p;
  };
  const size_t NODES = 65536, E = 524288;
  float* xbuf   = (float*)carve(NODES * 512 * 4);
  float* ybuf   = (float*)carve(NODES * 512 * 4);
  float* aggbuf = (float*)carve(NODES * 512 * 4);
  int* srcA = (int*)carve(E * 4); int* dstA = (int*)carve(E * 4); float* emA = (float*)carve(E * 4);
  int* srcB = (int*)carve(E * 4); int* dstB = (int*)carve(E * 4); float* emB = (float*)carve(E * 4);
  float* hsum = (float*)carve(1024 * 1024 * 4);
  float* h1 = (float*)carve(1024 * 512 * 4);
  float* h2 = (float*)carve(1024 * 256 * 4);
  float* h3 = (float*)carve(1024 * 128 * 4);
  float* h4 = (float*)carve(1024 * 64 * 4);
  // packed weight tiles: (K/32)*ntiles*2048 DWORDs each
  unsigned* w1l_p = (unsigned*)carve(16 * 4 * 2048 * 4);
  unsigned* w1r_p = (unsigned*)carve(16 * 4 * 2048 * 4);
  unsigned* w2l_p = (unsigned*)carve(16 * 4 * 2048 * 4);
  unsigned* w2r_p = (unsigned*)carve(16 * 4 * 2048 * 4);
  unsigned* w3l_p = (unsigned*)carve(16 * 4 * 2048 * 4);
  unsigned* w3r_p = (unsigned*)carve(16 * 4 * 2048 * 4);
  unsigned* lw1_p = (unsigned*)carve(32 * 4 * 2048 * 4);
  unsigned* lw2_p = (unsigned*)carve(16 * 2 * 2048 * 4);
  unsigned* lw3_p = (unsigned*)carve( 8 * 1 * 2048 * 4);
  unsigned* lw4_p = (unsigned*)carve( 4 * 1 * 2048 * 4);
  float* pn = (float*)carve(16);

  auto pack = [&](const float* W, unsigned* Wp, int K, int N) {
    int ntiles = (N + 127) >> 7;
    int total = (K >> 5) * ntiles * 2048;
    pack_w<<<(total + 255) / 256, 256, 0, stream>>>(W, Wp, K, N, ntiles);
  };
  pack(w1l, w1l_p, 512, 512); pack(w1r, w1r_p, 512, 512);
  pack(w2l, w2l_p, 512, 512); pack(w2r, w2r_p, 512, 512);
  pack(w3l, w3l_p, 512, 512); pack(w3r, w3r_p, 512, 512);
  pack(lw1, lw1_p, 1024, 512);
  pack(lw2, lw2_p, 512, 256);
  pack(lw3, lw3_p, 256, 128);
  pack(lw4, lw4_p, 128, 64);
  vec_norm<<<1, 256, 0, stream>>>(p1, pn + 0);
  vec_norm<<<1, 256, 0, stream>>>(p2, pn + 1);
  vec_norm<<<1, 256, 0, stream>>>(p3, pn + 2);

  embed_gather<<<65536, 128, 0, stream>>>(x_ids, emb, xbuf);

  // ---- level 1 (n_per=64 -> k=52)
  sage_agg<<<1024, 256, 0, stream>>>(xbuf, e_src, e_dst, nullptr, 64, aggbuf);
  gemm_wmma<<<dim3(4, 512), 256, 0, stream>>>(aggbuf, w1l_p, xbuf, w1r_p, b1l, ybuf, 65536, 512, 512, 1);
  topk_pool<<<1024, 256, 0, stream>>>(ybuf, 64, 52, p1, pn + 0, e_src, e_dst, nullptr, xbuf, srcA, dstA, emA);
  readout<<<1024, 256, 0, stream>>>(xbuf, 52, hsum, 0);

  // ---- level 2 (52 -> 42)
  sage_agg<<<1024, 256, 0, stream>>>(xbuf, srcA, dstA, emA, 52, aggbuf);
  gemm_wmma<<<dim3(4, 416), 256, 0, stream>>>(aggbuf, w2l_p, xbuf, w2r_p, b2l, ybuf, 53248, 512, 512, 1);
  topk_pool<<<1024, 256, 0, stream>>>(ybuf, 52, 42, p2, pn + 1, srcA, dstA, emA, xbuf, srcB, dstB, emB);
  readout<<<1024, 256, 0, stream>>>(xbuf, 42, hsum, 1);

  // ---- level 3 (42 -> 34)
  sage_agg<<<1024, 256, 0, stream>>>(xbuf, srcB, dstB, emB, 42, aggbuf);
  gemm_wmma<<<dim3(4, 336), 256, 0, stream>>>(aggbuf, w3l_p, xbuf, w3r_p, b3l, ybuf, 43008, 512, 512, 1);
  topk_pool<<<1024, 256, 0, stream>>>(ybuf, 42, 34, p3, pn + 2, srcB, dstB, emB, xbuf, srcA, dstA, emA);
  readout<<<1024, 256, 0, stream>>>(xbuf, 34, hsum, 1);

  // ---- MLP head
  gemm_wmma<<<dim3(4, 8), 256, 0, stream>>>(hsum, lw1_p, nullptr, nullptr, lb1, h1, 1024, 512, 1024, 1);
  gemm_wmma<<<dim3(2, 8), 256, 0, stream>>>(h1,   lw2_p, nullptr, nullptr, lb2, h2, 1024, 256, 512, 1);
  gemm_wmma<<<dim3(1, 8), 256, 0, stream>>>(h2,   lw3_p, nullptr, nullptr, lb3, h3, 1024, 128, 256, 1);
  gemm_wmma<<<dim3(1, 8), 256, 0, stream>>>(h3,   lw4_p, nullptr, nullptr, lb4, h4, 1024, 64, 128, 1);
  mlp_final<<<128, 256, 0, stream>>>(h4, lw5, lb5, (float*)d_out);
}
